// ScaleDotProductAttention_19164144075326
// MI455X (gfx1250) — matrix operand order repaired and checked by reading, not compiled, not run on hardware
//
#include <hip/hip_runtime.h>
#include <hip/hip_bf16.h>

// ---- CDNA5 WMMA types ------------------------------------------------------
typedef __attribute__((ext_vector_type(16))) __bf16 v16bf;
typedef __attribute__((ext_vector_type(8)))  __bf16 bf16x8;
typedef __attribute__((ext_vector_type(4)))  __bf16 bf16x4;
typedef __attribute__((ext_vector_type(8)))  float  v8f;
typedef __attribute__((ext_vector_type(4)))  float  f32x4;

#define SEQ 2048
#define DK  128
#define NELEM ((size_t)2 * 16 * SEQ * DK)   // elements per tensor (B*H*S*D)

// Load 8 consecutive elements as bf16x8. For float sources this uses
// __builtin_convertvector so the backend can emit packed v_cvt_pk_bf16_f32.
template <typename T>
static __device__ __forceinline__ bf16x8 ld8(const T* p) {
    if constexpr (sizeof(T) == 2) {
        return *(const bf16x8*)p;
    } else {
        f32x4 x = *(const f32x4*)p;
        f32x4 y = *(const f32x4*)(p + 4);
        bf16x4 a = __builtin_convertvector(x, bf16x4);
        bf16x4 b = __builtin_convertvector(y, bf16x4);
        return __builtin_shufflevector(a, b, 0, 1, 2, 3, 4, 5, 6, 7);
    }
}

template <typename T>
static __device__ __forceinline__ v16bf ld16(const T* plo, const T* phi) {
    bf16x8 lo = ld8(plo);
    bf16x8 hi = ld8(phi);
    return __builtin_shufflevector(lo, hi, 0, 1, 2, 3, 4, 5, 6, 7,
                                           8, 9, 10, 11, 12, 13, 14, 15);
}

// ============================================================================
// Pre-pass: f32 -> bf16 straight copy (Q, K), vectorized 8 elems/thread
// ============================================================================
__global__ __launch_bounds__(256) void cvt_bf16_kernel(
    const float* __restrict__ in, __bf16* __restrict__ out, long n8)
{
    size_t i = (size_t)blockIdx.x * 256 + threadIdx.x;
    if (i < (size_t)n8) {
        *(bf16x8*)(out + i * 8) = ld8(in + i * 8);
    }
}

// ============================================================================
// Pre-pass: V [B*H, S, DK] f32 -> Vt [B*H, DK, S] bf16 (transpose per head)
// One thread handles 8 consecutive k for a fixed (bh, d): 16B coalesced write.
// ============================================================================
__global__ __launch_bounds__(256) void cvt_tr_kernel(
    const float* __restrict__ V, __bf16* __restrict__ Vt)
{
    size_t idx = (size_t)blockIdx.x * 256 + threadIdx.x; // BH*DK*(S/8) items
    int k8 = idx & (SEQ / 8 - 1);
    int d  = (idx >> 8) & (DK - 1);
    int bh = (int)(idx >> 15);

    const float* src = V + ((size_t)bh * SEQ + (size_t)k8 * 8) * DK + d;
    bf16x8 v;
    #pragma unroll
    for (int j = 0; j < 8; ++j) v[j] = (__bf16)src[(size_t)j * DK];
    *(bf16x8*)(Vt + ((size_t)bh * DK + d) * SEQ + (size_t)k8 * 8) = v;
}

// ============================================================================
// Kernel 1: attn = softmax(mask(Q K^T * scale)) -> [B*H, S, S]
// grid = (S/16, B*H), block = 256 (8 waves). Wave w owns keys [w*256, +256).
// ============================================================================
template <typename T>
__global__ __launch_bounds__(256) void attn_scores_kernel(
    const T* __restrict__ Q, const T* __restrict__ K,
    const int* __restrict__ mask, float* __restrict__ attn,
    int H, float scale)
{
    const int lane = threadIdx.x & 31;
    const int wid  = threadIdx.x >> 5;          // 0..7
    const int bh   = blockIdx.y;                // 0..B*H-1
    const int b    = bh / H;
    const int q0   = blockIdx.x * 16;

    const int  mrow = lane & 15;                // M (A) or N (B/C) index
    const bool hi   = lane >= 16;

    const T* Qbase = Q + ((size_t)bh * SEQ + q0) * DK;
    const T* Kbase = K + (size_t)bh * SEQ * DK;

    // A fragments: Q tile 16x128 as 4x (16x32 bf16), ISA A-layout.
    // lanes 0-15: M=lane, K={0..7,16..23}; lanes 16-31: M=lane-16, K={8..15,24..31}
    v16bf a[4];
    {
        const T* qrow = Qbase + (size_t)mrow * DK;
        #pragma unroll
        for (int i = 0; i < 4; ++i)
            a[i] = ld16(qrow + i * 32 + (hi ? 8 : 0),
                        qrow + i * 32 + (hi ? 24 : 16));
    }

    const int kw0 = wid * 256;                  // this wave's first key

    // QK^T: 16 column tiles, 4 WMMAs each over D=128
    v8f st[16];
    #pragma unroll
    for (int kt = 0; kt < 16; ++kt) {
        // warm GL2 for the K tile two steps ahead (global_prefetch_b8)
        if (kt + 2 < 16)
            __builtin_prefetch(Kbase + (size_t)(kw0 + (kt + 2) * 16 + mrow) * DK, 0, 1);

        v8f c = {};
        // B fragment (32x16): N = mrow (key within tile), K-dim = d (ISA layout, matches A)
        const T* ktile = Kbase + (size_t)(kw0 + kt * 16 + mrow) * DK;
        #pragma unroll
        for (int i = 0; i < 4; ++i) {
            const T* kp = ktile + i * 32 + (hi ? 8 : 0);
            v16bf bfrag = ld16(kp, kp + 16);
            c = __builtin_amdgcn_wmma_f32_16x16x32_bf16(
                    false, a[i], false, bfrag, (short)0, c, false, false);
        }
        st[kt] = c;
    }

    // scale + mask. C layout: VGPR r -> lanes 0-15: M=r; lanes 16-31: M=8+r
    const int* mbase = mask + ((size_t)b * SEQ + q0) * SEQ;
    #pragma unroll
    for (int kt = 0; kt < 16; ++kt) {
        const int kcol = kw0 + kt * 16 + mrow;
        #pragma unroll
        for (int r = 0; r < 8; ++r) {
            const int row = r + (hi ? 8 : 0);
            float s = st[kt][r] * scale;
            int   m = mbase[(size_t)row * SEQ + kcol];
            st[kt][r] = (m == 0) ? -1e9f : s;
        }
    }

    // row max: tiles -> lanes (xor within 16-lane half) -> waves via LDS
    __shared__ float red[8][16];
    float gmax[8];
    #pragma unroll
    for (int r = 0; r < 8; ++r) {
        float m = st[0][r];
        #pragma unroll
        for (int kt = 1; kt < 16; ++kt) m = fmaxf(m, st[kt][r]);
        #pragma unroll
        for (int off = 1; off < 16; off <<= 1)
            m = fmaxf(m, __shfl_xor(m, off, 32));
        if (lane == 0 || lane == 16) red[wid][r + (hi ? 8 : 0)] = m;
    }
    __syncthreads();
    #pragma unroll
    for (int r = 0; r < 8; ++r) {
        const int row = r + (hi ? 8 : 0);
        float m = red[0][row];
        #pragma unroll
        for (int w = 1; w < 8; ++w) m = fmaxf(m, red[w][row]);
        gmax[r] = m;
    }
    __syncthreads();

    // exp + row sum
    float rinv[8];
    #pragma unroll
    for (int r = 0; r < 8; ++r) {
        float ssum = 0.f;
        #pragma unroll
        for (int kt = 0; kt < 16; ++kt) {
            float e = __expf(st[kt][r] - gmax[r]);
            st[kt][r] = e;
            ssum += e;
        }
        #pragma unroll
        for (int off = 1; off < 16; off <<= 1)
            ssum += __shfl_xor(ssum, off, 32);
        if (lane == 0 || lane == 16) red[wid][r + (hi ? 8 : 0)] = ssum;
    }
    __syncthreads();
    #pragma unroll
    for (int r = 0; r < 8; ++r) {
        const int row = r + (hi ? 8 : 0);
        float s = 0.f;
        #pragma unroll
        for (int w = 0; w < 8; ++w) s += red[w][row];
        rinv[r] = 1.0f / s;
    }

    // normalize + write attn
    float* abase = attn + ((size_t)bh * SEQ + q0) * SEQ;
    #pragma unroll
    for (int kt = 0; kt < 16; ++kt) {
        const int kcol = kw0 + kt * 16 + mrow;
        #pragma unroll
        for (int r = 0; r < 8; ++r) {
            const int row = r + (hi ? 8 : 0);
            abase[(size_t)row * SEQ + kcol] = st[kt][r] * rinv[r];
        }
    }
}

// ============================================================================
// Kernel 2: context = attn @ V -> [B*H, S, DK]
// T == __bf16: V is transposed bf16 [B*H, DK, S] (contiguous B-fragments)
// T == float : V is original f32 [B*H, S, DK] (strided gather + convert)
// ============================================================================
template <typename T>
__global__ __launch_bounds__(256) void attn_context_kernel(
    const float* __restrict__ attn, const T* __restrict__ V,
    float* __restrict__ ctx)
{
    const int lane = threadIdx.x & 31;
    const int wid  = threadIdx.x >> 5;
    const int bh   = blockIdx.y;
    const int q0   = blockIdx.x * 16;

    const int  mrow = lane & 15;
    const bool hi   = lane >= 16;

    __shared__ float acc[16 * DK];
    for (int i = threadIdx.x; i < 16 * DK; i += 256) acc[i] = 0.f;
    __syncthreads();

    const float* Abase = attn + ((size_t)bh * SEQ + q0) * SEQ;   // P: [16, S]

    v8f c[8];
    #pragma unroll
    for (int d = 0; d < 8; ++d) c[d] = (v8f){};

    const int kw0 = wid * 256;
    #pragma unroll 1
    for (int kp = 0; kp < 8; ++kp) {            // 8 chunks of 32 keys
        const int kbase = kw0 + kp * 32;

        // A fragment: P tile 16x32 (rows contiguous in k), f32 -> bf16 packed
        const float* arow = Abase + (size_t)mrow * SEQ + kbase;
        __builtin_prefetch(arow + 32, 0, 1);    // next P chunk -> global_prefetch_b8
        v16bf afrag = ld16(arow + (hi ? 16 : 0), arow + (hi ? 24 : 8));

        #pragma unroll
        for (int d = 0; d < 8; ++d) {
            // B fragment: V 32x16, N = mrow (d within tile), K = key
            v16bf bfrag;
            if constexpr (sizeof(T) == 2) {
                // Vt[bh][d][k]: 16 contiguous bf16 per lane
                const T* vrow = V + ((size_t)bh * DK + d * 16 + mrow) * SEQ
                                  + kbase + (hi ? 16 : 0);
                bfrag = ld16(vrow, vrow + 8);
            } else {
                const T* vcol = V + (size_t)bh * SEQ * DK
                                  + (size_t)(kbase + (hi ? 16 : 0)) * DK
                                  + d * 16 + mrow;
                #pragma unroll
                for (int e = 0; e < 16; ++e) bfrag[e] = (__bf16)vcol[(size_t)e * DK];
            }
            c[d] = __builtin_amdgcn_wmma_f32_16x16x32_bf16(
                       false, afrag, false, bfrag, (short)0, c[d], false, false);
        }
    }

    // cross-wave reduction via LDS float atomics (ds_add_f32)
    #pragma unroll
    for (int d = 0; d < 8; ++d) {
        #pragma unroll
        for (int r = 0; r < 8; ++r) {
            const int row = r + (hi ? 8 : 0);
            atomicAdd(&acc[row * DK + d * 16 + mrow], c[d][r]);
        }
    }
    __syncthreads();

    float* out = ctx + ((size_t)bh * SEQ + q0) * DK;
    for (int i = threadIdx.x; i < 16 * DK; i += 256) out[i] = acc[i];
}

// ============================================================================
extern "C" void kernel_launch(void* const* d_in, const int* in_sizes, int n_in,
                              void* d_out, int out_size, void* d_ws, size_t ws_size,
                              hipStream_t stream) {
    (void)in_sizes; (void)n_in; (void)out_size;
    const int B = 2, H = 16;
    const float scale = 0.08838834764831843f;   // 1/sqrt(128)

    const float* Q    = (const float*)d_in[0];
    const float* K    = (const float*)d_in[1];
    const float* V    = (const float*)d_in[2];
    const int*   mask = (const int*)d_in[3];

    float* ctx  = (float*)d_out;                                // [B,H,S,DK]
    float* attn = (float*)d_out + NELEM;                        // [B,H,S,S]

    dim3 grid(SEQ / 16, B * H);
    dim3 block(256);

    if (ws_size >= NELEM * 2 * 3) {
        // Fast path: one-time bf16 conversion (Q, K row-major; V transposed)
        __bf16* Qb = (__bf16*)d_ws;
        __bf16* Kb = Qb + NELEM;
        __bf16* Vt = Kb + NELEM;

        const int nCvtBlk = (int)(NELEM / 8 / 256);             // 4096
        cvt_bf16_kernel<<<nCvtBlk, 256, 0, stream>>>(Q, Qb, (long)(NELEM / 8));
        cvt_bf16_kernel<<<nCvtBlk, 256, 0, stream>>>(K, Kb, (long)(NELEM / 8));
        cvt_tr_kernel<<<nCvtBlk, 256, 0, stream>>>(V, Vt);

        attn_scores_kernel<__bf16><<<grid, block, 0, stream>>>(Qb, Kb, mask, attn, H, scale);
        attn_context_kernel<__bf16><<<grid, block, 0, stream>>>(attn, Vt, ctx);
    } else {
        // Fallback: convert inside the kernels
        attn_scores_kernel<float><<<grid, block, 0, stream>>>(Q, K, mask, attn, H, scale);
        attn_context_kernel<float><<<grid, block, 0, stream>>>(attn, V, ctx);
    }
}
